// LinearAttention_17609365914142
// MI455X (gfx1250) — matrix operand order repaired and checked by reading, not compiled, not run on hardware
//
#include <hip/hip_runtime.h>

// ---------- builtin vector types (no HIP_vector_type through AS1!) ----------
typedef __attribute__((ext_vector_type(4))) unsigned int u32x4;
typedef __attribute__((ext_vector_type(2))) unsigned int u32x2;
typedef __attribute__((ext_vector_type(4))) float        f32x4;
typedef __attribute__((ext_vector_type(16))) __bf16 v16bf;
typedef __attribute__((ext_vector_type(8)))  float  v8f;

// ---------- global-address-space pointer types ----------
#define AS1 __attribute__((address_space(1)))
typedef const __bf16 AS1*  gc_bf;
typedef __bf16 AS1*        g_bf;
typedef const float AS1*   gc_f32;
typedef float AS1*         g_f32;
typedef const u32x2 AS1*   gc_u2;
typedef u32x2 AS1*         g_u2;
typedef const u32x4 AS1*   gc_u4;
typedef const f32x4 AS1*   gc_f4;
// Generic->global cast (exact for global pointers on amdgcn).
#define GPTR(T, p) ((T)(unsigned long long)(p))

#define EPSV 1e-6f

// ---------- scalar helpers ----------
static __device__ __forceinline__ unsigned short f2bf_bits(float f) {
  unsigned int u = __builtin_bit_cast(unsigned int, f);
  u = (u + 0x7FFFu + ((u >> 16) & 1u)) >> 16;  // round-to-nearest-even
  return (unsigned short)u;
}
static __device__ __forceinline__ __bf16 f2bf(float f) {
  return __builtin_bit_cast(__bf16, f2bf_bits(f));
}
static __device__ __forceinline__ float bits2f(unsigned int b16) {
  unsigned int u = b16 << 16;
  return __builtin_bit_cast(float, u);
}
static __device__ __forceinline__ float elu1(float x) {
  return x > 0.0f ? (x + 1.0f) : __expf(x);   // elu(x)+1
}

// ---------- WMMA fragment loaders (contiguous-k order, same for A and B) ----------
// A 16x32 bf16: lane holds row (lane&15); 16 consecutive K starting at
// (lane>>4)*16.  p = &A[row][k0 + (lane>>4)*16].
static __device__ __forceinline__ v16bf ldfragA(gc_bf p) {
  union { v16bf v; u32x4 u[2]; } r;
  r.u[0] = *(gc_u4)(p);
  r.u[1] = *(gc_u4)(p + 8);
  return r.v;
}
// B 32x16 bf16 (row-major Bw[N,K], computing A*Bw^T): lane holds column
// (lane&15); 16 consecutive K starting at (lane>>4)*16.
static __device__ __forceinline__ v16bf ldfragB(gc_bf p) {
  union { v16bf v; u32x4 u[2]; } r;
  r.u[0] = *(gc_u4)(p);
  r.u[1] = *(gc_u4)(p + 8);
  return r.v;
}

// ---------- f32 -> bf16 convert (x4 packed) ----------
__global__ void cvt_f32_bf16_x4(const float* __restrict__ src,
                                unsigned int* __restrict__ dst, long n4) {
  long i = (long)blockIdx.x * blockDim.x + threadIdx.x;
  if (i < n4) {
    f32x4 f = GPTR(gc_f4, src)[i];
    u32x2 o;
    o.x = ((unsigned int)f2bf_bits(f.y) << 16) | f2bf_bits(f.x);
    o.y = ((unsigned int)f2bf_bits(f.w) << 16) | f2bf_bits(f.z);
    GPTR(g_u2, dst)[i] = o;
  }
}

// =====================================================================
// BIG GEMM: C = A[M,K] * Bw[Nn,K]^T, 8 waves as 2x4, wave tile 64x64
// (16 WMMAs per 16 b128 loads per k-step). Dims must tile exactly:
// M % 128 == 0, Nn % 256 == 0, K % 32 == 0.
// MODE 0: QKV epilogue (elu+1 on q,k; scatter q->[BH,N,D], k,v->[BH,D,N])
// MODE 3: proj epilogue (add bias, f32 out)
// =====================================================================
template <int MODE>
__global__ __launch_bounds__(256) void gemm_wmma_big(
    const __bf16* __restrict__ A, const __bf16* __restrict__ Bw,
    int lda, int ldb, int M, int Nn, int K,
    __bf16* __restrict__ q_ws, __bf16* __restrict__ kT_ws,
    __bf16* __restrict__ vT_ws,
    const float* __restrict__ bias, float* __restrict__ outf) {
  const int lane = threadIdx.x & 31;
  const int wave = threadIdx.x >> 5;
  const int wm = wave >> 2;  // 0..1
  const int wn = wave & 3;   // 0..3
  const int Rbase = blockIdx.y * 128 + wm * 64;
  const int Cbase = blockIdx.x * 256 + wn * 64;

  const int l15   = lane & 15;
  const int koffA = (lane >> 4) * 16;
  const int koffB = (lane >> 4) * 16;

  gc_bf Ab = GPTR(gc_bf, A);
  gc_bf Bb = GPTR(gc_bf, Bw);
  int offA[4], offB[4];
#pragma unroll
  for (int i = 0; i < 4; ++i) {
    offA[i] = (Rbase + i * 16 + l15) * lda + koffA;
    offB[i] = (Cbase + i * 16 + l15) * ldb + koffB;
  }

  v8f acc[4][4] = {};

  for (int k0 = 0; k0 < K; k0 += 32) {
    if (k0 + 512 < K) {
      __builtin_prefetch(A + offA[0] + k0 + 512, 0, 3);
      __builtin_prefetch(A + offA[2] + k0 + 512, 0, 3);
      __builtin_prefetch(Bw + offB[0] + k0 + 512, 0, 3);
      __builtin_prefetch(Bw + offB[2] + k0 + 512, 0, 3);
    }
    v16bf bfr[4];
#pragma unroll
    for (int j = 0; j < 4; ++j) bfr[j] = ldfragB(Bb + offB[j] + k0);
#pragma unroll
    for (int i = 0; i < 4; ++i) {
      v16bf afr = ldfragA(Ab + offA[i] + k0);
#pragma unroll
      for (int j = 0; j < 4; ++j)
        acc[i][j] = __builtin_amdgcn_wmma_f32_16x16x32_bf16(
            false, afr, false, bfr[j], (short)0, acc[i][j], false, false);
    }
  }

  // C/D layout: n = lane&15 ; m = r + (lane>>4)*8 for element r of v8f.
  const int hi8 = (lane >> 4) * 8;
#pragma unroll
  for (int i = 0; i < 4; ++i) {
#pragma unroll
    for (int j = 0; j < 4; ++j) {
      const int R0 = Rbase + i * 16;
      const int C0 = Cbase + j * 16;
#pragma unroll
      for (int r = 0; r < 8; ++r) {
        float val = acc[i][j][r];
        int m = R0 + hi8 + r;
        int f = C0 + l15;
        if (MODE == 0) {
          int b = m >> 12, n = m & 4095;
          int which = f >> 10, h = (f >> 6) & 15, d = f & 63;
          long bh2 = (long)(b * 16 + h);
          if (which == 0)
            GPTR(g_bf, q_ws)[(bh2 * 4096 + n) * 64 + d] = f2bf(elu1(val));
          else if (which == 1)
            GPTR(g_bf, kT_ws)[(bh2 * 64 + d) * 4096 + n] = f2bf(elu1(val));
          else
            GPTR(g_bf, vT_ws)[(bh2 * 64 + d) * 4096 + n] = f2bf(val);
        } else {
          GPTR(g_f32, outf)[(long)m * 1024 + f] = val + GPTR(gc_f32, bias)[f];
        }
      }
    }
  }
}

// =====================================================================
// SMALL GEMM (per-head, batched over grid.z): wave tile 32x32,
// 8 waves as 4x2, block tile 128x64.
// MODE 1: kv epilogue  (store transposed bf16 kvT[bh][e][d])
// MODE 2: attn epilogue (scale by zrow, scatter bf16 to [B,N,C])
// =====================================================================
template <int MODE>
__global__ __launch_bounds__(256) void gemm_wmma_small(
    const __bf16* __restrict__ A, const __bf16* __restrict__ Bw,
    int lda, int ldb, int M, int Nn, int K,
    long aBatch, long bBatch,
    __bf16* __restrict__ kvT_ws,
    const float* __restrict__ zrow, __bf16* __restrict__ attn_ws) {
  const int lane = threadIdx.x & 31;
  const int wave = threadIdx.x >> 5;
  const int wm = wave >> 1;  // 0..3
  const int wn = wave & 1;   // 0..1
  const int bh = blockIdx.z;
  A  += (long)bh * aBatch;
  Bw += (long)bh * bBatch;

  const int Rbase = blockIdx.y * 128 + wm * 32;
  const int Cbase = blockIdx.x * 64 + wn * 32;
  if (Rbase >= M || Cbase >= Nn) return;  // wave-uniform -> EXEC stays all-ones

  const int l15   = lane & 15;
  const int koffA = (lane >> 4) * 16;
  const int koffB = (lane >> 4) * 16;

  gc_bf Ab = GPTR(gc_bf, A);
  gc_bf Bb = GPTR(gc_bf, Bw);
  const int oA0 = (Rbase + l15) * lda + koffA;
  const int oA1 = oA0 + 16 * lda;
  const int oB0 = (Cbase + l15) * ldb + koffB;
  const int oB1 = oB0 + 16 * ldb;

  v8f acc00 = {}; v8f acc01 = {}; v8f acc10 = {}; v8f acc11 = {};

  for (int k0 = 0; k0 < K; k0 += 32) {
    if (k0 + 256 < K) {
      __builtin_prefetch(A + oA0 + k0 + 256, 0, 3);
      __builtin_prefetch(Bw + oB0 + k0 + 256, 0, 3);
    }
    v16bf a0 = ldfragA(Ab + oA0 + k0);
    v16bf a1 = ldfragA(Ab + oA1 + k0);
    v16bf b0 = ldfragB(Bb + oB0 + k0);
    v16bf b1 = ldfragB(Bb + oB1 + k0);
    acc00 = __builtin_amdgcn_wmma_f32_16x16x32_bf16(false, a0, false, b0,
                                                    (short)0, acc00, false, false);
    acc01 = __builtin_amdgcn_wmma_f32_16x16x32_bf16(false, a0, false, b1,
                                                    (short)0, acc01, false, false);
    acc10 = __builtin_amdgcn_wmma_f32_16x16x32_bf16(false, a1, false, b0,
                                                    (short)0, acc10, false, false);
    acc11 = __builtin_amdgcn_wmma_f32_16x16x32_bf16(false, a1, false, b1,
                                                    (short)0, acc11, false, false);
  }

  const int hi8 = (lane >> 4) * 8;
  auto epilogue = [&](v8f acc, int R0, int C0) {
#pragma unroll
    for (int r = 0; r < 8; ++r) {
      float val = acc[r];
      int m = R0 + hi8 + r;
      int f = C0 + l15;
      if (MODE == 1) {
        GPTR(g_bf, kvT_ws)[((long)bh * 64 + f) * 64 + m] = f2bf(val);
      } else {
        float z = GPTR(gc_f32, zrow)[(long)bh * 4096 + m];
        int b = bh >> 4, h = bh & 15;
        GPTR(g_bf, attn_ws)[((long)b * 4096 + m) * 1024 + h * 64 + f] =
            f2bf(val * z);
      }
    }
  };
  epilogue(acc00, Rbase,      Cbase);
  epilogue(acc01, Rbase,      Cbase + 16);
  epilogue(acc10, Rbase + 16, Cbase);
  epilogue(acc11, Rbase + 16, Cbase + 16);
}

// ---------- ksum[bh*64+d] = sum_n kT[bh][d][n]  (wave per row) ----------
__global__ __launch_bounds__(256) void ksum_kernel(const __bf16* __restrict__ kT,
                                                   float* __restrict__ ksum) {
  int row  = blockIdx.x * (blockDim.x >> 5) + (threadIdx.x >> 5);  // 0..4095
  int lane = threadIdx.x & 31;
  const __bf16* p = kT + (long)row * 4096;
  float s = 0.0f;
  for (int i = lane * 4; i < 4096; i += 128) {
    u32x2 u = *GPTR(gc_u2, p + i);
    s += bits2f(u.x & 0xFFFFu) + bits2f(u.x >> 16) +
         bits2f(u.y & 0xFFFFu) + bits2f(u.y >> 16);
  }
  for (int o = 16; o > 0; o >>= 1) s += __shfl_down(s, o, 32);
  if (lane == 0) GPTR(g_f32, ksum)[row] = s;
}

// ---------- zrow[bh*4096+n] = 1/(q[bh,n,:] . ksum[bh,:] + eps) ----------
__global__ __launch_bounds__(256) void zrow_kernel(const __bf16* __restrict__ q,
                                                   const float* __restrict__ ksum,
                                                   float* __restrict__ z) {
  long i = (long)blockIdx.x * blockDim.x + threadIdx.x;  // 0..262143
  long bhi = i >> 12;
  const __bf16* qp = q + i * 64;
  gc_f32 kp = GPTR(gc_f32, ksum) + bhi * 64;
  float s = 0.0f;
#pragma unroll
  for (int d = 0; d < 64; d += 4) {
    u32x2 u = *GPTR(gc_u2, qp + d);
    s += bits2f(u.x & 0xFFFFu) * kp[d + 0];
    s += bits2f(u.x >> 16)     * kp[d + 1];
    s += bits2f(u.y & 0xFFFFu) * kp[d + 2];
    s += bits2f(u.y >> 16)     * kp[d + 3];
  }
  GPTR(g_f32, z)[i] = 1.0f / (s + EPSV);
}

// ---------- host launcher ----------
extern "C" void kernel_launch(void* const* d_in, const int* in_sizes, int n_in,
                              void* d_out, int out_size, void* d_ws, size_t ws_size,
                              hipStream_t stream) {
  const float* x     = (const float*)d_in[0];  // [4,4096,1024]
  const float* Wqkv  = (const float*)d_in[1];  // [3072,1024]
  const float* Wproj = (const float*)d_in[2];  // [1024,1024]
  const float* bproj = (const float*)d_in[3];  // [1024]
  float* out = (float*)d_out;                  // [4,4096,1024]

  char* base = (char*)d_ws;
  size_t off = 0;
  auto carve = [&](size_t bytes) -> char* {
    char* p = base + off;
    off += (bytes + 255) & ~(size_t)255;
    return p;
  };
  __bf16* x_bf   = (__bf16*)carve(33554432);  // 16M bf16
  __bf16* wq_bf  = (__bf16*)carve(6291456);
  __bf16* wp_bf  = (__bf16*)carve(2097152);
  __bf16* q_ws   = (__bf16*)carve(33554432);  // [64,4096,64]
  __bf16* kT_ws  = (__bf16*)carve(33554432);  // [64,64,4096]
  __bf16* vT_ws  = (__bf16*)carve(33554432);  // [64,64,4096]
  __bf16* kvT_ws = (__bf16*)carve(524288);    // [64,64,64]
  float*  ksum   = (float*)carve(16384);      // [64,64]
  float*  zrow   = (float*)carve(1048576);    // [64,4096]
  __bf16* attn_ws = x_bf;  // alias: x_bf is dead after the QKV GEMM

  // 1) f32 -> bf16 conversions
  cvt_f32_bf16_x4<<<16384, 256, 0, stream>>>(x,     (unsigned int*)x_bf, 4194304);
  cvt_f32_bf16_x4<<<3072,  256, 0, stream>>>(Wqkv,  (unsigned int*)wq_bf, 786432);
  cvt_f32_bf16_x4<<<1024,  256, 0, stream>>>(Wproj, (unsigned int*)wp_bf, 262144);

  // 2) QKV GEMM: [16384,1024] x [3072,1024]^T, feature-map + scatter epilogue
  gemm_wmma_big<0><<<dim3(12, 128, 1), 256, 0, stream>>>(
      x_bf, wq_bf, 1024, 1024, 16384, 3072, 1024,
      q_ws, kT_ws, vT_ws, nullptr, nullptr);

  // 3) ksum and z
  ksum_kernel<<<512, 256, 0, stream>>>(kT_ws, ksum);
  zrow_kernel<<<1024, 256, 0, stream>>>(q_ws, ksum, zrow);

  // 4) kv[bh] = kT[64,4096] x vT[64,4096]^T  (K=N=4096), store transposed
  gemm_wmma_small<1><<<dim3(1, 1, 64), 256, 0, stream>>>(
      kT_ws, vT_ws, 4096, 4096, 64, 64, 4096,
      (long)64 * 4096, (long)64 * 4096, kvT_ws, nullptr, nullptr);

  // 5) attn[bh] = q[4096,64] x kvT[64,64]^T, scale by z, scatter to [B,N,C]
  gemm_wmma_small<2><<<dim3(1, 32, 64), 256, 0, stream>>>(
      q_ws, kvT_ws, 64, 64, 4096, 64, 64,
      (long)4096 * 64, (long)64 * 64, nullptr, zrow, attn_ws);

  // 6) proj GEMM: [16384,1024] x [1024,1024]^T + bias -> f32 out
  gemm_wmma_big<3><<<dim3(4, 128, 1), 256, 0, stream>>>(
      attn_ws, wp_bf, 1024, 1024, 16384, 1024, 1024,
      nullptr, nullptr, nullptr, bproj, out);
}
